// MultiHeadAttention_764504179347
// MI455X (gfx1250) — compile-verified, matrix operand order checked by reading, not run on hardware
//
#include <hip/hip_runtime.h>

// ---------------------------------------------------------------------------
// CDNA5 (gfx1250) fused MHA:
//   1) one-time fp32->bf16 casts (X flat, W transposed)
//   2) projection GEMMs: TDM (tensor_load_to_lds) double-buffered tiles + WMMA
//   3) fused streaming attention with WMMA + per-wave LDS P-tile relayout
// ---------------------------------------------------------------------------

typedef __attribute__((ext_vector_type(16))) __bf16    bf16x16;
typedef __attribute__((ext_vector_type(8)))  float     f32x8;
typedef __attribute__((ext_vector_type(4)))  unsigned  v4u;
typedef __attribute__((ext_vector_type(8)))  int       v8i;
typedef __attribute__((ext_vector_type(4)))  int       v4i;

// One 16x32 bf16 WMMA operand slice: 16 bf16 per lane = 8 VGPRs.
struct Frag32 { uint4 lo, hi; };

__device__ __forceinline__ unsigned short f2bf(float f) {
  unsigned u = __builtin_bit_cast(unsigned, f);
  u += 0x7FFFu + ((u >> 16) & 1u);          // round-to-nearest-even
  return (unsigned short)(u >> 16);
}
__device__ __forceinline__ unsigned pack2bf(float a, float b) {
  return (unsigned)f2bf(a) | ((unsigned)f2bf(b) << 16);
}

__device__ __forceinline__ f32x8 wmma_bf16(const Frag32& a, const Frag32& b, f32x8 c) {
  return __builtin_amdgcn_wmma_f32_16x16x32_bf16(
      false, __builtin_bit_cast(bf16x16, a),
      false, __builtin_bit_cast(bf16x16, b),
      (short)0, c, false, false);
}

// ---------------------------------------------------------------------------
// TDM: load a 64-row x 32-col bf16 tile (row stride 1024 elems) into LDS with
// 16 bytes of padding after every 64-byte row (LDS row stride 80 B == [.][40]).
// D# per ISA 8.3-8.6: count=1, type=2, data_size=2B, pad_interval=16dw(+4dw).
// ---------------------------------------------------------------------------
__device__ __forceinline__ void tdm_tile_64x32(unsigned lds_off,
                                               const unsigned short* gp,
                                               unsigned tensor_dim1) {
  unsigned long long ga = (unsigned long long)(size_t)gp;
  v4u g0;
  g0.x = 1u;                                        // count=1 (valid user D#)
  g0.y = lds_off;                                   // lds_addr
  g0.z = (unsigned)(ga & 0xFFFFFFFFu);              // global_addr[31:0]
  g0.w = (unsigned)((ga >> 32) & 0x01FFFFFFu) | (2u << 30);  // [56:32] | type=2
  v8i g1;
  g1[0] = (int)((1u << 16) |                        // data_size = 2 bytes
                (1u << 20) |                        // pad_enable
                (3u << 22) |                        // pad_interval: 16 DWORDs
                (3u << 25));                        // pad_amount:   4 DWORDs
  g1[1] = (int)((1024u & 0xFFFFu) << 16);           // tensor_dim0[15:0]
  g1[2] = (int)((1024u >> 16) | ((tensor_dim1 & 0xFFFFu) << 16));
  g1[3] = (int)((tensor_dim1 >> 16) | (32u << 16)); // tile_dim0 = 32
  g1[4] = (int)64;                                  // tile_dim1 = 64 (dim2 = 0)
  g1[5] = (int)1024;                                // tensor_dim0_stride
  g1[6] = 0;
  g1[7] = 0;
  v4i z4 = {0, 0, 0, 0};
#if defined(__clang_major__) && (__clang_major__ >= 23)
  v8i z8 = {0, 0, 0, 0, 0, 0, 0, 0};
  __builtin_amdgcn_tensor_load_to_lds(g0, g1, z4, z4, z8, 0);
#else
  __builtin_amdgcn_tensor_load_to_lds(g0, g1, z4, z4, 0);
#endif
}

// ---------------------------------------------------------------------------
// One-time casts: fp32 -> bf16 (8 elems/thread), and W fp32 -> Wt bf16 [n][k].
// ---------------------------------------------------------------------------
__global__ __launch_bounds__(256)
void cast_bf16x8(const float* __restrict__ src, unsigned short* __restrict__ dst,
                 int n8) {
  int i = blockIdx.x * 256 + threadIdx.x;
  if (i < n8) {
    const float4* s = (const float4*)src + (size_t)i * 2;
    float4 a = s[0], b = s[1];
    uint4 o;
    o.x = pack2bf(a.x, a.y); o.y = pack2bf(a.z, a.w);
    o.z = pack2bf(b.x, b.y); o.w = pack2bf(b.z, b.w);
    ((uint4*)dst)[i] = o;
  }
}

__global__ __launch_bounds__(256)
void cast_transpose_w(const float* __restrict__ W, unsigned short* __restrict__ Wt) {
  __shared__ unsigned short t[64][65];
  const int tid = threadIdx.x;
  const int kb = blockIdx.y * 64, nb = blockIdx.x * 64;
  #pragma unroll
  for (int i = 0; i < 16; ++i) {
    int idx = tid + i * 256;
    int r = idx >> 6, c = idx & 63;                  // r = k, c = n (coalesced read)
    t[c][r] = f2bf(W[(size_t)(kb + r) * 1024 + nb + c]);
  }
  __syncthreads();
  #pragma unroll
  for (int i = 0; i < 16; ++i) {
    int idx = tid + i * 256;
    int r = idx >> 6, c = idx & 63;                  // r = n, c = k (coalesced write)
    Wt[(size_t)(nb + r) * 1024 + kb + c] = t[r][c];
  }
}

// ---------------------------------------------------------------------------
// Projection: out = (Xb @ Wt^T + bias) [* 0.125 for Q]; bf16 in, bf16 out.
//   Xb: [8192,1024] bf16 ; Wt: [1024 n][1024 k] bf16 ; bias: [1024] f32
//   mode 0/1 (Q/K): out[b,h,s,d] ; mode 2 (V): out[b,h,d,s] (transposed)
// TDM double-buffered tile staging; 4 wave32 per block; 64x64 output tile.
// ---------------------------------------------------------------------------
__global__ __launch_bounds__(128, 2)
void proj_tdm_wmma(const unsigned short* __restrict__ Xb,
                   const unsigned short* __restrict__ Wt,
                   const float* __restrict__ bias,
                   unsigned short* __restrict__ out, int mode) {
  __shared__ __align__(16) unsigned short Asm[2][64][40];  // TDM-padded rows
  __shared__ __align__(16) unsigned short Bsm[2][64][40];

  const int tid  = threadIdx.x;
  const int wv   = tid >> 5, lane = tid & 31;
  const int g    = lane >> 4, lr = lane & 15;
  const int h    = blockIdx.x;                 // N tile == head (64 == head dim)
  const int nbase = h * 64;
  const int mbase = blockIdx.y * 64;

  const unsigned a0 = (unsigned)(size_t)&Asm[0][0][0];
  const unsigned a1 = (unsigned)(size_t)&Asm[1][0][0];
  const unsigned b0 = (unsigned)(size_t)&Bsm[0][0][0];
  const unsigned b1 = (unsigned)(size_t)&Bsm[1][0][0];

  if (wv == 0) {                               // prologue: chunk 0 into buffer 0
    tdm_tile_64x32(a0, Xb + (size_t)mbase * 1024, 8192u);
    tdm_tile_64x32(b0, Wt + (size_t)nbase * 1024, 1024u);
  }

  f32x8 acc[4] = {};
  int buf = 0;
  for (int kk = 0; kk < 1024; kk += 32) {
    if (wv == 0) {
      if (kk + 32 < 1024) {                    // issue next chunk into other buffer
        tdm_tile_64x32(buf ? a0 : a1, Xb + (size_t)mbase * 1024 + kk + 32, 8192u);
        tdm_tile_64x32(buf ? b0 : b1, Wt + (size_t)nbase * 1024 + kk + 32, 1024u);
        __builtin_amdgcn_s_wait_tensorcnt(2);  // oldest 2 (current chunk) done
      } else {
        __builtin_amdgcn_s_wait_tensorcnt(0);
      }
    }
    __syncthreads();

    const int arow = wv * 16 + lr;
    Frag32 a;
    a.lo = *(const uint4*)&Asm[buf][arow][g * 8];
    a.hi = *(const uint4*)&Asm[buf][arow][16 + g * 8];
    #pragma unroll
    for (int nt = 0; nt < 4; ++nt) {
      Frag32 b;
      b.lo = *(const uint4*)&Bsm[buf][nt * 16 + lr][g * 8];
      b.hi = *(const uint4*)&Bsm[buf][nt * 16 + lr][16 + g * 8];
      acc[nt] = wmma_bf16(a, b, acc[nt]);
    }
    __syncthreads();
    buf ^= 1;
  }

  const float scale = (mode == 0) ? 0.125f : 1.0f;   // fold 1/sqrt(64) into q
  #pragma unroll
  for (int nt = 0; nt < 4; ++nt) {
    const int d = nt * 16 + lr;
    const float bv = bias[nbase + d];
    #pragma unroll
    for (int r = 0; r < 8; ++r) {
      const int m = mbase + wv * 16 + r + 8 * g;     // C layout: row r+8*(lane/16)
      const int b = m >> 10, s = m & 1023;
      const float val = (acc[nt][r] + bv) * scale;
      if (mode < 2)
        out[(((size_t)(b * 16 + h) * 1024 + s) << 6) + d] = f2bf(val);
      else
        out[(((size_t)(b * 16 + h) * 64 + d) << 10) + s] = f2bf(val);
    }
  }
}

// ---------------------------------------------------------------------------
// Fused attention per (b,h): scores = exp(q k^T) (q pre-scaled), denom = rowsum,
// ctx = (scores @ v) / (denom + 1e-8). Streams keys in chunks of 32.
//   qb/kb: [B*H, 1024, 64] bf16 ; vt: [B*H, 64, 1024] bf16 ; out: [B,1024,1024] f32
// ---------------------------------------------------------------------------
__global__ __launch_bounds__(128, 2)
void attn_fused_wmma(const unsigned short* __restrict__ qb,
                     const unsigned short* __restrict__ kb,
                     const unsigned short* __restrict__ vt,
                     float* __restrict__ out) {
  __shared__ __align__(16) unsigned short stage[4][16][40];  // per-wave 16x32 P tile

  const int tid = threadIdx.x;
  const int wv  = tid >> 5, lane = tid & 31;
  const int g   = lane >> 4, lr = lane & 15;
  const int bh  = blockIdx.y;
  const int qrow = blockIdx.x * 64 + wv * 16 + lr;

  const unsigned short* qp = qb + ((size_t)(bh * 1024 + qrow) << 6);
  Frag32 a_lo, a_hi;
  a_lo.lo = *(const uint4*)(qp + g * 8);
  a_lo.hi = *(const uint4*)(qp + 16 + g * 8);
  a_hi.lo = *(const uint4*)(qp + 32 + g * 8);
  a_hi.hi = *(const uint4*)(qp + 48 + g * 8);

  f32x8 ctx[4] = {};
  float rs[8] = {};

  for (int j = 0; j < 1024; j += 32) {
    // prefetch next key/value chunk into cache (global_prefetch_b8)
    if (j + 32 < 1024) {
      __builtin_prefetch(kb + ((size_t)(bh * 1024 + j + 32 + lr) << 6), 0, 1);
      __builtin_prefetch(vt + ((size_t)(bh * 64 + lr) << 10) + j + 32, 0, 1);
    }
    // --- scores for two 16-key subtiles; exp; stage as bf16 A-layout 16x32 ---
    #pragma unroll
    for (int t = 0; t < 2; ++t) {
      const unsigned short* kp = kb + ((size_t)(bh * 1024 + j + t * 16 + lr) << 6);
      Frag32 k_lo, k_hi;
      k_lo.lo = *(const uint4*)(kp + g * 8);
      k_lo.hi = *(const uint4*)(kp + 16 + g * 8);
      k_hi.lo = *(const uint4*)(kp + 32 + g * 8);
      k_hi.hi = *(const uint4*)(kp + 48 + g * 8);
      f32x8 c = {};
      c = wmma_bf16(a_lo, k_lo, c);
      c = wmma_bf16(a_hi, k_hi, c);
      #pragma unroll
      for (int r = 0; r < 8; ++r) {
        float e = __expf(c[r]);
        rs[r] += e;                                   // per-lane column partials
        stage[wv][r + 8 * g][t * 16 + lr] = f2bf(e);  // C-layout -> staged tile
      }
    }
    asm volatile("s_wait_dscnt 0" ::: "memory");      // order wave-local LDS RAW

    Frag32 p;
    p.lo = *(const uint4*)&stage[wv][lr][g * 8];
    p.hi = *(const uint4*)&stage[wv][lr][16 + g * 8];

    #pragma unroll
    for (int nt = 0; nt < 4; ++nt) {
      const unsigned short* vp = vt + ((size_t)(bh * 64 + nt * 16 + lr) << 10) + j;
      Frag32 vb;
      vb.lo = *(const uint4*)(vp + g * 8);
      vb.hi = *(const uint4*)(vp + 16 + g * 8);
      ctx[nt] = wmma_bf16(p, vb, ctx[nt]);
    }
  }

  #pragma unroll
  for (int r = 0; r < 8; ++r) {
    float v = rs[r];
    v += __shfl_xor(v, 1, 32);
    v += __shfl_xor(v, 2, 32);
    v += __shfl_xor(v, 4, 32);
    v += __shfl_xor(v, 8, 32);
    rs[r] = 1.0f / (v + 1e-8f);
  }

  const int b = bh >> 4, h = bh & 15;
  #pragma unroll
  for (int nt = 0; nt < 4; ++nt) {
    const int d = h * 64 + nt * 16 + lr;
    #pragma unroll
    for (int r = 0; r < 8; ++r) {
      const int s = blockIdx.x * 64 + wv * 16 + r + 8 * g;
      out[(size_t)(b * 1024 + s) * 1024 + d] = ctx[nt][r] * rs[r];
    }
  }
}

// ---------------------------------------------------------------------------
extern "C" void kernel_launch(void* const* d_in, const int* in_sizes, int n_in,
                              void* d_out, int out_size, void* d_ws, size_t ws_size,
                              hipStream_t stream) {
  (void)in_sizes; (void)n_in; (void)out_size; (void)ws_size;
  const float* Q  = (const float*)d_in[0];
  const float* K  = (const float*)d_in[1];
  const float* V  = (const float*)d_in[2];
  const float* Wq = (const float*)d_in[3];
  const float* bq = (const float*)d_in[4];
  const float* Wk = (const float*)d_in[5];
  const float* bk = (const float*)d_in[6];
  const float* Wv = (const float*)d_in[7];
  const float* bv = (const float*)d_in[8];

  const size_t elems = (size_t)8 * 16 * 1024 * 64;   // 8.4M bf16 per S x 1024 buf
  unsigned short* qbuf = (unsigned short*)d_ws;      // [B,H,S,64]
  unsigned short* kbuf = qbuf + elems;               // [B,H,S,64]
  unsigned short* vtb  = kbuf + elems;               // [B,H,64,S]
  unsigned short* xbf  = vtb  + elems;               // [8192,1024] bf16 (reused x3)
  unsigned short* wtb  = xbf  + elems;               // [1024,1024]  bf16 (reused x3)

  const int n8 = (int)(elems / 8);
  dim3 gc(n8 / 256 + (n8 % 256 != 0)), bc(256);
  dim3 gw(16, 16), bw(256);
  dim3 gp(16, 128), bp(128);

  // serialized on stream: cast -> proj for each of Q/K/V (buffers reused)
  cast_bf16x8<<<gc, bc, 0, stream>>>(Q, xbf, n8);
  cast_transpose_w<<<gw, bw, 0, stream>>>(Wq, wtb);
  proj_tdm_wmma<<<gp, bp, 0, stream>>>(xbf, wtb, bq, qbuf, 0);

  cast_bf16x8<<<gc, bc, 0, stream>>>(K, xbf, n8);
  cast_transpose_w<<<gw, bw, 0, stream>>>(Wk, wtb);
  proj_tdm_wmma<<<gp, bp, 0, stream>>>(xbf, wtb, bk, kbuf, 1);

  cast_bf16x8<<<gc, bc, 0, stream>>>(V, xbf, n8);
  cast_transpose_w<<<gw, bw, 0, stream>>>(Wv, wtb);
  proj_tdm_wmma<<<gp, bp, 0, stream>>>(xbf, wtb, bv, vtb, 2);

  attn_fused_wmma<<<dim3(16, 128), dim3(128), 0, stream>>>(qbuf, kbuf, vtb,
                                                           (float*)d_out);
}
